// AttentionHead_11751030522642
// MI455X (gfx1250) — matrix operand order repaired and checked by reading, not compiled, not run on hardware
//
#include <hip/hip_runtime.h>
#include <hip/hip_bf16.h>

typedef _Float16 h16;
typedef __attribute__((ext_vector_type(16))) _Float16 v16h;
typedef __attribute__((ext_vector_type(8)))  float    v8f;

#define WMMA_F16(a, b, c) \
  __builtin_amdgcn_wmma_f32_16x16x32_f16(false, (a), false, (b), (short)0, (c), false, false)

union frag16 { v16h v; uint4 u[2]; };

// ---------------------------------------------------------------------------
// Kernel 0a: f32 -> f16 conversion, 8 elements per thread (b128 in, b128 out)
// ---------------------------------------------------------------------------
__global__ __launch_bounds__(256)
void cvt_f32_to_f16(const float* __restrict__ in, h16* __restrict__ out) {
  size_t i = (size_t)blockIdx.x * 256 + threadIdx.x;
  const float4* p = (const float4*)in + i * 2;
  float4 a = p[0];
  float4 b = p[1];
  union { h16 h[8]; uint4 u; } t;
  t.h[0] = (h16)a.x; t.h[1] = (h16)a.y; t.h[2] = (h16)a.z; t.h[3] = (h16)a.w;
  t.h[4] = (h16)b.x; t.h[5] = (h16)b.y; t.h[6] = (h16)b.z; t.h[7] = (h16)b.w;
  ((uint4*)out)[i] = t.u;
}

// ---------------------------------------------------------------------------
// Kernel 0b: W [1024][1024] f32  ->  Wt [n][k] f16  (LDS tile transpose)
// Wt row-major-in-k is exactly the WMMA B-operand friendly layout.
// ---------------------------------------------------------------------------
__global__ __launch_bounds__(1024)
void transpose_w_f16(const float* __restrict__ W, h16* __restrict__ Wt) {
  __shared__ float tile[32][33];
  int tx = threadIdx.x, ty = threadIdx.y;
  int k = blockIdx.y * 32 + ty;
  int n = blockIdx.x * 32 + tx;
  tile[ty][tx] = W[(size_t)k * 1024 + n];
  __syncthreads();
  int nO = blockIdx.x * 32 + ty;
  int kO = blockIdx.y * 32 + tx;
  Wt[(size_t)nO * 1024 + kO] = (h16)tile[tx][ty];
}

// ---------------------------------------------------------------------------
// Kernel 1: Y[M=8192][N=1024] = A[M][K=1024] @ Bt[N][K]^T, f16 in, f32 accum.
// 256 threads = 8 waves; workgroup tile 128x128; wave tile 32x64 = 2x4 WMMA.
// Fragments loaded straight from global as b128 (layout matches ISA operand
// maps): A lane l: row=l&15, K chunks {8*(l/16)..+8} and {16+8*(l/16)..+8};
// B lane l: col=l&15, K chunks {8*(l/16)..+8} and {16+8*(l/16)..+8}.
// out_mode 0: Y[row][col] f16 (scaled). out_mode 1: V-transpose [b][col][s].
// ---------------------------------------------------------------------------
__global__ __launch_bounds__(256)
void gemm_f16_wmma(const h16* __restrict__ A, const h16* __restrict__ Bt,
                   h16* __restrict__ Y, float scale, int out_mode) {
  const int K = 1024, N = 1024;
  int tid  = threadIdx.x;
  int lane = tid & 31, w = tid >> 5;
  int wm = w & 3, wn = w >> 2;
  int m0 = blockIdx.y * 128 + wm * 32;
  int n0 = blockIdx.x * 128 + wn * 64;
  int c = lane & 15, hi = lane >> 4;

  v8f acc[2][4];
  for (int i = 0; i < 2; i++)
    for (int j = 0; j < 4; j++)
      for (int r = 0; r < 8; r++) acc[i][j][r] = 0.f;

  for (int k = 0; k < K; k += 32) {
    frag16 af[2];
#pragma unroll
    for (int i = 0; i < 2; i++) {
      const h16* ap = A + (size_t)(m0 + i * 16 + c) * K + k + hi * 8;
      af[i].u[0] = *(const uint4*)(ap);
      af[i].u[1] = *(const uint4*)(ap + 16);
    }
    frag16 bf[4];
#pragma unroll
    for (int j = 0; j < 4; j++) {
      const h16* bp = Bt + (size_t)(n0 + j * 16 + c) * K + k + hi * 8;
      bf[j].u[0] = *(const uint4*)(bp);
      bf[j].u[1] = *(const uint4*)(bp + 16);
    }
#pragma unroll
    for (int i = 0; i < 2; i++)
#pragma unroll
      for (int j = 0; j < 4; j++)
        acc[i][j] = WMMA_F16(af[i].v, bf[j].v, acc[i][j]);
  }

#pragma unroll
  for (int i = 0; i < 2; i++)
#pragma unroll
    for (int j = 0; j < 4; j++)
#pragma unroll
      for (int r = 0; r < 8; r++) {
        int row = m0 + i * 16 + hi * 8 + r;
        int col = n0 + j * 16 + c;
        float v = acc[i][j][r] * scale;
        if (out_mode == 0) {
          Y[(size_t)row * N + col] = (h16)v;
        } else {  // V transposed: [b][d=col][s], b = row/2048, s = row%2048
          int b = row >> 11, s = row & 2047;
          Y[((size_t)b * N + col) * 2048 + s] = (h16)v;
        }
      }
}

// ---------------------------------------------------------------------------
// Kernel 2: causal flash attention. Q has log2(e)/sqrt(d) pre-folded, so all
// exponentials are exp2 (single v_exp_f32 each).
// Workgroup: 32 query rows (blockIdx.x), batch = blockIdx.y.
// 8 waves: each holds the full 32-row x 128-col slice of O in registers
// (16 tiles x v8f = 128 VGPRs). Per 64-wide k-block:
//   (1) 8 waves each do one 16x16 score sub-tile (K=1024, 32 WMMAs) -> LDS
//   (2) online softmax, 8 threads/row + shfl_xor tree reductions -> P f16 LDS
//   (3) rescale O, then O += P @ V  (V fragments direct from global Vt)
// ---------------------------------------------------------------------------
__global__ __launch_bounds__(256)
void attn_f16_wmma(const h16* __restrict__ Q, const h16* __restrict__ Kx,
                   const h16* __restrict__ Vt, float* __restrict__ out) {
  const int D = 1024, S = 2048;
  const int QB = 32, KBL = 64;
  __shared__ float Sld[QB * KBL];     // 8 KB score tile (f32, log2-scaled)
  __shared__ h16   Pld[QB * KBL];     // 4 KB prob tile (f16)
  __shared__ float mS[QB], lS[QB], corrS[QB];

  int b  = blockIdx.y;
  int q0 = blockIdx.x * QB;
  int tid = threadIdx.x, lane = tid & 31, w = tid >> 5;
  int c = lane & 15, hi = lane >> 4;

  const h16* Qb = Q  + (size_t)b * S * D;
  const h16* Kb = Kx + (size_t)b * S * D;
  const h16* Vb = Vt + (size_t)b * D * S;   // [d][s]

  if (tid < QB) { mS[tid] = -__builtin_huge_valf(); lS[tid] = 0.f; }

  v8f acc[2][8];
  for (int i = 0; i < 2; i++)
    for (int j = 0; j < 8; j++)
      for (int r = 0; r < 8; r++) acc[i][j][r] = 0.f;

  int dW = w * 128;            // this wave's output column slice
  int tm = w & 1, tn = w >> 1; // this wave's score sub-tile

  // softmax work split: 8 threads per row, 8 columns per thread
  int srow = tid >> 3;         // 0..31
  int sseg = tid & 7;          // 0..7

  int kbEnd = (q0 + QB + KBL - 1) / KBL;  // causal: only k-blocks <= diagonal
  for (int kb = 0; kb < kbEnd; ++kb) {
    int k0 = kb * KBL;

    // ---- score sub-tile: S = Q[q0+16tm .. +16, :] @ K[k0+16tn .. +16, :]^T
    v8f s8;
    for (int r = 0; r < 8; r++) s8[r] = 0.f;
    for (int d0 = 0; d0 < D; d0 += 32) {
      frag16 a, bb;
      const h16* ap = Qb + (size_t)(q0 + tm * 16 + c) * D + d0 + hi * 8;
      a.u[0] = *(const uint4*)ap;
      a.u[1] = *(const uint4*)(ap + 16);
      const h16* bp = Kb + (size_t)(k0 + tn * 16 + c) * D + d0 + hi * 8;
      bb.u[0] = *(const uint4*)bp;
      bb.u[1] = *(const uint4*)(bp + 16);
      s8 = WMMA_F16(a.v, bb.v, s8);
    }

    __syncthreads();  // B1: prior iter's Pld/corrS reads done; safe to write Sld
#pragma unroll
    for (int r = 0; r < 8; r++) {
      int qrow = tm * 16 + hi * 8 + r;
      int kcol = k0 + tn * 16 + c;
      float v = s8[r];
      if (kcol > q0 + qrow) v = -__builtin_huge_valf();  // causal mask
      Sld[qrow * KBL + tn * 16 + c] = v;
    }
    __syncthreads();  // B2

    // ---- online softmax (base 2): 8 threads per row, lanewise reductions.
    // The 8 threads of a row occupy 8 consecutive lanes of one wave, so
    // shfl_xor with masks 1/2/4 reduces within the row group (wave32-safe).
    {
      float sv[8];
      float pmax = -__builtin_huge_valf();
#pragma unroll
      for (int j = 0; j < 8; j++) {
        sv[j] = Sld[srow * KBL + sseg * 8 + j];
        pmax = fmaxf(pmax, sv[j]);
      }
      pmax = fmaxf(pmax, __shfl_xor(pmax, 1, 32));
      pmax = fmaxf(pmax, __shfl_xor(pmax, 2, 32));
      pmax = fmaxf(pmax, __shfl_xor(pmax, 4, 32));
      float mOld = mS[srow];
      float mNew = fmaxf(mOld, pmax);
      float rs = 0.f;
#pragma unroll
      for (int j = 0; j < 8; j++) {
        float p = __builtin_amdgcn_exp2f(sv[j] - mNew);  // v_exp_f32
        rs += p;
        Pld[srow * KBL + sseg * 8 + j] = (h16)p;
      }
      rs += __shfl_xor(rs, 1, 32);
      rs += __shfl_xor(rs, 2, 32);
      rs += __shfl_xor(rs, 4, 32);
      if (sseg == 0) {
        float corr = __builtin_amdgcn_exp2f(mOld - mNew);
        mS[srow] = mNew;
        lS[srow] = lS[srow] * corr + rs;
        corrS[srow] = corr;
      }
    }
    __syncthreads();  // B3

    // ---- P fragments from LDS (A-operand layout; reused across 8 D tiles)
    frag16 pf[2][2];
#pragma unroll
    for (int tA = 0; tA < 2; tA++)
#pragma unroll
      for (int ks = 0; ks < 2; ks++) {
        const h16* pp = Pld + (tA * 16 + c) * KBL + ks * 32 + hi * 8;
        pf[tA][ks].u[0] = *(const uint4*)pp;
        pf[tA][ks].u[1] = *(const uint4*)(pp + 16);
      }

    // ---- rescale O by 2^(m_old - m_new), then O += P @ V
#pragma unroll
    for (int tA = 0; tA < 2; tA++) {
      float cr[8];
#pragma unroll
      for (int r = 0; r < 8; r++) cr[r] = corrS[tA * 16 + hi * 8 + r];
#pragma unroll
      for (int tB = 0; tB < 8; tB++) {
#pragma unroll
        for (int r = 0; r < 8; r++) acc[tA][tB][r] *= cr[r];
#pragma unroll
        for (int ks = 0; ks < 2; ks++) {
          frag16 vf;  // B operand: V[s][d], K-dim = s contiguous in Vt rows
          const h16* vp = Vb + (size_t)(dW + tB * 16 + c) * S + k0 + ks * 32 + hi * 8;
          vf.u[0] = *(const uint4*)vp;
          vf.u[1] = *(const uint4*)(vp + 16);
          acc[tA][tB] = WMMA_F16(pf[tA][ks].v, vf.v, acc[tA][tB]);
        }
      }
    }
  }

  // ---- finalize: O /= l, write f32 output [b][q][d]
  float* ob = out + (size_t)b * S * D;
#pragma unroll
  for (int tA = 0; tA < 2; tA++)
#pragma unroll
    for (int tB = 0; tB < 8; tB++)
#pragma unroll
      for (int r = 0; r < 8; r++) {
        int qrow = tA * 16 + hi * 8 + r;
        int col  = dW + tB * 16 + c;
        ob[(size_t)(q0 + qrow) * D + col] = acc[tA][tB][r] / lS[qrow];
      }
}

// ---------------------------------------------------------------------------
extern "C" void kernel_launch(void* const* d_in, const int* in_sizes, int n_in,
                              void* d_out, int out_size, void* d_ws, size_t ws_size,
                              hipStream_t stream) {
  (void)in_sizes; (void)n_in; (void)out_size; (void)ws_size;
  const float* Xk = (const float*)d_in[0];  // inputs_for_keys
  const float* Xv = (const float*)d_in[1];  // inputs_for_values
  const float* Xq = (const float*)d_in[2];  // inputs_for_queries
  const float* Wk = (const float*)d_in[3];
  const float* Wv = (const float*)d_in[4];
  const float* Wq = (const float*)d_in[5];
  float* out = (float*)d_out;

  const size_t NE = (size_t)4 * 2048 * 1024;  // 8M elems per activation tensor
  const size_t MM = (size_t)1024 * 1024;      // 1M elems per weight
  h16* ws  = (h16*)d_ws;
  h16* bXk = ws;                       // converted X tensors
  h16* bXv = ws + NE;
  h16* bXq = ws + 2 * NE;
  h16* wKt = ws + 3 * NE;              // transposed f16 weights
  h16* wVt = ws + 3 * NE + MM;
  h16* wQt = ws + 3 * NE + 2 * MM;
  h16* bQ  = ws + 3 * NE + 3 * MM;     // Q projection (pre-scaled)
  h16* bK  = bXq;                      // K reuses Xq buffer (free after Q GEMM)
  h16* bVt = bXk;                      // Vt reuses Xk buffer (free after K GEMM)
  // total workspace: 4*16MB + 3*2MB = 70 MB

  dim3 cvtGrid((unsigned)(NE / (256 * 8)));
  cvt_f32_to_f16<<<cvtGrid, 256, 0, stream>>>(Xk, bXk);
  cvt_f32_to_f16<<<cvtGrid, 256, 0, stream>>>(Xv, bXv);
  cvt_f32_to_f16<<<cvtGrid, 256, 0, stream>>>(Xq, bXq);

  dim3 tb(32, 32), tg(32, 32);
  transpose_w_f16<<<tg, tb, 0, stream>>>(Wk, wKt);
  transpose_w_f16<<<tg, tb, 0, stream>>>(Wv, wVt);
  transpose_w_f16<<<tg, tb, 0, stream>>>(Wq, wQt);

  // Q scale: 1/sqrt(1024) * log2(e)  -> softmax becomes pure exp2
  const float qscale = 1.44269504088896340736f / 32.0f;

  dim3 gg(1024 / 128, 8192 / 128);  // (8, 64)
  gemm_f16_wmma<<<gg, 256, 0, stream>>>(bXq, wQt, bQ, qscale, 0);
  gemm_f16_wmma<<<gg, 256, 0, stream>>>(bXk, wKt, bK, 1.0f, 0);
  gemm_f16_wmma<<<gg, 256, 0, stream>>>(bXv, wVt, bVt, 1.0f, 1);  // V transposed

  dim3 ag(2048 / 32, 4);  // (64 q-blocks, 4 batches)
  attn_f16_wmma<<<ag, 256, 0, stream>>>(bQ, bK, bVt, out);
}
